// FastWeightPoseMemory_31009663877361
// MI455X (gfx1250) — matrix operand order repaired and checked by reading, not compiled, not run on hardware
//
#include <hip/hip_runtime.h>
#include <cstdint>
#include <cstddef>

// ---------------- CDNA5 WMMA plumbing (gfx1250, wave32) ----------------
typedef __attribute__((ext_vector_type(16))) __bf16 v16bf;
typedef __attribute__((ext_vector_type(8)))  float  v8f;

union FragAB { v16bf v; unsigned int u[8]; };
union FragC  { v8f   v; float f[8]; };

#if __has_builtin(__builtin_amdgcn_tensor_load_to_lds) && __has_builtin(__builtin_amdgcn_s_wait_tensorcnt)
#define HAVE_TDM 1
typedef __attribute__((ext_vector_type(4))) unsigned int v4u;
typedef __attribute__((ext_vector_type(8))) int v8i;
typedef __attribute__((ext_vector_type(4))) int v4i;
#endif

__device__ __forceinline__ unsigned short f2b(float f) {
    union { float f; unsigned u; } c; c.f = f;
    unsigned u = c.u;
    return (unsigned short)((u + 0x7FFFu + ((u >> 16) & 1u)) >> 16);
}
__device__ __forceinline__ float b2f(unsigned short b) {
    union { unsigned u; float f; } c; c.u = ((unsigned)b) << 16; return c.f;
}

__device__ __forceinline__ v8f wmma_bf16(v16bf a, v16bf b, v8f c) {
    return __builtin_amdgcn_wmma_f32_16x16x32_bf16(
        false, a, false, b, (short)0, c, false, false);
}

// A-fragment 16x32 bf16 from row-major LDS tile (ld in halfs): 2x ds_load_b128
__device__ __forceinline__ v16bf load_frag_a(const unsigned short* lds, int ld) {
    int lane = threadIdx.x & 31;
    const unsigned short* p = lds + (lane & 15) * ld + ((lane >> 4) << 4);
    FragAB a;
#pragma unroll
    for (int j = 0; j < 4; ++j) {
        a.u[j]     = *(const unsigned int*)(p + (j << 1));
        a.u[j + 4] = *(const unsigned int*)(p + 8 + (j << 1));
    }
    return a.v;
}

// Packed-fragment load: lane reads its contiguous 32B chunk (2x ds_load_b128)
__device__ __forceinline__ v16bf load_frag_packed(const unsigned short* base) {
    const unsigned int* p = (const unsigned int*)(base + (threadIdx.x & 31) * 16);
    FragAB f;
#pragma unroll
    for (int j = 0; j < 8; ++j) f.u[j] = p[j];
    return f.v;
}

#ifdef HAVE_TDM
// Issue one TDM 2D tile load: rows x rowElems (bf16) from row-major global
// (stride strideElems) into LDS at ldsOff, LDS rows padded by padAmount DWORDs
// after every padInterval DWORDs (codes per ISA D# group1).
// Call from ONE wave; follow with s_wait_tensorcnt + block barrier.
__device__ __forceinline__ void tdm_load_2d(const unsigned short* gsrc, unsigned ldsOff,
                                            int rowElems, int rows, int strideElems,
                                            int padIntervalCode, int padAmountCode) {
    unsigned long long ga = (unsigned long long)(uintptr_t)gsrc;
    v4u g0;
    g0.x = 1u;                                    // count=1, no gather
    g0.y = ldsOff;                                // lds_addr (bytes)
    g0.z = (unsigned)ga;                          // global_addr[31:0]
    g0.w = (unsigned)((ga >> 32) & 0x01FFFFFFu) | (2u << 30);  // addr[56:32] | type=2
    v8i g1;
    g1[0] = (1 << 16) | (1 << 20) |               // data_size=2B, pad_enable
            (padIntervalCode << 22) | (padAmountCode << 25);
    g1[1] = (int)((strideElems & 0xFFFF) << 16);  // tensor_dim0 low16 (bits 63:48)
    g1[2] = (int)(((strideElems >> 16) & 0xFFFF) | ((rows & 0xFFFF) << 16));
    g1[3] = (int)(((rows >> 16) & 0xFFFF) | ((rowElems & 0xFFFF) << 16)); // tile_dim0
    g1[4] = rows & 0xFFFF;                        // tile_dim1
    g1[5] = strideElems;                          // tensor_dim0_stride low32
    g1[6] = 0; g1[7] = 0;
    v4i gz4 = (v4i)0;                             // groups 2/3 unused (2D tensor)
    v8i gz8 = (v8i)0;                             // trailing group (clang-23 6-arg form)
    __builtin_amdgcn_tensor_load_to_lds(g0, g1, gz4, gz4, gz8, 0);
}
#endif

// B packing: tile [K x N], element (k,n):
//   idx = ((k>>5)*(N/16)+(n>>4))*512 + ((((k>>4)&1)<<4)+(n&15))*16 + (k&15)
__device__ __forceinline__ int packB_idx(int k, int n, int n16) {
    return (((k >> 5) * n16 + (n >> 4)) << 9) +
           (((((k >> 4) & 1) << 4) + (n & 15)) << 4) + (k & 15);
}

// ---------------- problem constants ----------------
#define NH 16
#define HD 64
#define NIN 1024
#define NOUT 1024
#define NBT 32768                 // B*T
static const size_t NELEM = 33554432ULL;   // NBT*NOUT

// ---------------- conversion kernels ----------------
__global__ void convert_kernel(const float* __restrict__ src,
                               unsigned short* __restrict__ dst, int n) {
    int i = blockIdx.x * 256 + threadIdx.x;
    if (i < n) dst[i] = f2b(src[i]);
}
__global__ void transpose_convert_kernel(const float* __restrict__ src,
                                         unsigned short* __restrict__ dst,
                                         int rows, int cols) {
    int b = blockIdx.y;
    int i = blockIdx.x * 256 + threadIdx.x;
    if (i >= rows * cols) return;
    int r = i / cols, c = i % cols;
    size_t base = (size_t)b * rows * cols;
    dst[base + (size_t)c * rows + r] = f2b(src[base + i]);
}
__global__ void zerof_kernel(float* p, int n) {
    int i = blockIdx.x * 256 + threadIdx.x;
    if (i < n) p[i] = 0.f;
}

// ---------------- generic big GEMM: C[M,N] = A[M,K] @ B[K,N] (+bias) ----------------
__global__ __launch_bounds__(256)
void gemm_bf16_128x64(const unsigned short* __restrict__ A,
                      const unsigned short* __restrict__ B,
                      const float* __restrict__ bias,
                      float* __restrict__ outF,
                      unsigned short* __restrict__ outB,
                      int M, int N, int K) {
    __shared__ __align__(16) unsigned short As[128 * 40];
    __shared__ __align__(16) unsigned short Bs[4 * 512];   // packed 32x64
    int tid = threadIdx.x;
    int m0 = blockIdx.y * 128;
    int n0 = blockIdx.x * 64;
    int w = tid >> 5, lane = tid & 31;
    int wm = (w >> 1) * 32, wn = (w & 1) * 32;
    // fixed staging assignments
    int arr = tid >> 3, ac = (tid & 7) << 2;        // A: 4 rows (arr+32i), cols ac..ac+3
    int bk = tid >> 4, bn = (tid & 15) << 2;        // B: rows bk, bk+16; cols bn..bn+3
    FragC c[2][2];
#pragma unroll
    for (int i = 0; i < 2; ++i)
#pragma unroll
        for (int j = 0; j < 2; ++j)
#pragma unroll
            for (int r = 0; r < 8; ++r) c[i][j].f[r] = 0.f;

    for (int k0 = 0; k0 < K; k0 += 32) {
#ifdef HAVE_TDM
        if (tid < 32) {                            // wave 0 drives the Tensor Data Mover
            tdm_load_2d(A + (size_t)m0 * K + k0, (unsigned)(uintptr_t)As,
                        32, 128, K, /*pad every 16 DW*/3, /*pad 4 DW*/3);
        }
#else
#pragma unroll
        for (int i = 0; i < 4; ++i)
            *(uint2*)&As[(arr + (i << 5)) * 40 + ac] =
                *(const uint2*)(A + (size_t)(m0 + arr + (i << 5)) * K + k0 + ac);
#endif
#pragma unroll
        for (int i = 0; i < 2; ++i) {              // B tile 32x64 -> packed layout
            int k = bk + (i << 4);
            uint2 g = *(const uint2*)(B + (size_t)(k0 + k) * N + n0 + bn);
            unsigned short* d = &Bs[((bn >> 4) << 9) + (((i << 4) + (bn & 15)) << 4) + bk];
            d[0]  = (unsigned short)(g.x & 0xffff);
            d[16] = (unsigned short)(g.x >> 16);
            d[32] = (unsigned short)(g.y & 0xffff);
            d[48] = (unsigned short)(g.y >> 16);
        }
        if (k0 + 32 < K) {                         // prefetch next A k-tile
            __builtin_prefetch(A + (size_t)(m0 + (tid >> 1)) * K + k0 + 32, 0, 1);
        }
#ifdef HAVE_TDM
        if (tid < 32) __builtin_amdgcn_s_wait_tensorcnt(0);
#endif
        __syncthreads();
        v16bf a0 = load_frag_a(&As[wm * 40], 40);
        v16bf a1 = load_frag_a(&As[(wm + 16) * 40], 40);
        v16bf b0 = load_frag_packed(&Bs[(wn >> 4) << 9]);
        v16bf b1 = load_frag_packed(&Bs[((wn >> 4) + 1) << 9]);
        c[0][0].v = wmma_bf16(a0, b0, c[0][0].v);
        c[0][1].v = wmma_bf16(a0, b1, c[0][1].v);
        c[1][0].v = wmma_bf16(a1, b0, c[1][0].v);
        c[1][1].v = wmma_bf16(a1, b1, c[1][1].v);
        __syncthreads();
    }
#pragma unroll
    for (int i = 0; i < 2; ++i)
#pragma unroll
        for (int j = 0; j < 2; ++j) {
            int col = n0 + wn + j * 16 + (lane & 15);
            int rowb = m0 + wm + i * 16 + ((lane >> 4) << 3);
            float bv = bias ? bias[col] : 0.f;
#pragma unroll
            for (int r = 0; r < 8; ++r) {
                float v = c[i][j].f[r] + bv;
                size_t idx = (size_t)(rowb + r) * N + col;
                if (outF) outF[idx] = v;
                else      outB[idx] = f2b(v);
            }
        }
}

// stage a 64x64 row-major weight block into packed-B LDS (K=64, N=64)
__device__ __forceinline__ void stage_weight_packed(const unsigned short* __restrict__ g,
                                                    unsigned short* lds) {
    int tid = threadIdx.x;
    int nq = (tid & 15) << 2;
#pragma unroll
    for (int i = 0; i < 4; ++i) {
        int k = (tid >> 4) + (i << 4);             // 0..63
        uint2 d = *(const uint2*)(g + k * 64 + nq);
        unsigned short* p = &lds[((((k >> 5) << 2) + (nq >> 4)) << 9) +
                                 ((((k >> 4) & 1) << 4) + (nq & 15)) * 16 + (k & 15)];
        p[0]  = (unsigned short)(d.x & 0xffff);
        p[16] = (unsigned short)(d.x >> 16);
        p[32] = (unsigned short)(d.y & 0xffff);
        p[48] = (unsigned short)(d.y >> 16);
    }
}

// stage a 128-row x 64-col activation tile (ld 1024) into row-major LDS (ld 72)
__device__ __forceinline__ void stage_act_tile(const unsigned short* __restrict__ g,
                                               unsigned short* lds) {
#ifdef HAVE_TDM
    if (threadIdx.x < 32) {
        tdm_load_2d(g, (unsigned)(uintptr_t)lds, 64, 128, 1024,
                    /*pad every 32 DW*/4, /*pad 4 DW*/3);
        __builtin_amdgcn_s_wait_tensorcnt(0);
    }
#else
    int tid = threadIdx.x;
    int c4 = (tid & 15) << 2;
#pragma unroll
    for (int i = 0; i < 8; ++i) {
        int r = (tid >> 4) + (i << 4);
        *(uint2*)&lds[r * 72 + c4] = *(const uint2*)(g + (size_t)r * 1024 + c4);
    }
#endif
}

// ---------------- fused per-head forward ----------------
__global__ __launch_bounds__(256)
void head_forward_kernel(const unsigned short* __restrict__ P,
                         const unsigned short* __restrict__ W1,
                         const unsigned short* __restrict__ W2,
                         const unsigned short* __restrict__ W3,
                         unsigned short* __restrict__ U,
                         unsigned short* __restrict__ V,
                         unsigned short* __restrict__ SP,
                         unsigned short* __restrict__ S,
                         unsigned short* __restrict__ X) {
    __shared__ __align__(16) unsigned short pqs[128 * 72];
    __shared__ __align__(16) unsigned short w1s[8 * 512];  // packed 64x64
    __shared__ __align__(16) unsigned short w2s[8 * 512];
    __shared__ __align__(16) unsigned short w3s[8 * 512];
    __shared__ __align__(16) unsigned short sst[8 * 16 * 72];
    int tid = threadIdx.x, h = blockIdx.x, m0 = blockIdx.y * 128;
    stage_act_tile(P + (size_t)m0 * 1024 + h * 64, pqs);
    stage_weight_packed(W1 + h * 4096, w1s);
    stage_weight_packed(W2 + h * 4096, w2s);
    stage_weight_packed(W3 + h * 4096, w3s);
    __syncthreads();
    int w = tid >> 5, lane = tid & 31, rw = w * 16;
    FragC ca[4], cb[4];
#pragma unroll
    for (int nf = 0; nf < 4; ++nf)
#pragma unroll
        for (int r = 0; r < 8; ++r) { ca[nf].f[r] = 0.f; cb[nf].f[r] = 0.f; }
#pragma unroll
    for (int kt = 0; kt < 64; kt += 32) {
        v16bf a = load_frag_a(&pqs[rw * 72 + kt], 72);
        int kb = (kt >> 5) << 2;
#pragma unroll
        for (int nf = 0; nf < 4; ++nf) {
            ca[nf].v = wmma_bf16(a, load_frag_packed(&w1s[(kb + nf) << 9]), ca[nf].v);
            cb[nf].v = wmma_bf16(a, load_frag_packed(&w2s[(kb + nf) << 9]), cb[nf].v);
        }
    }
    unsigned short* ss = &sst[w * 16 * 72];
#pragma unroll
    for (int nf = 0; nf < 4; ++nf) {
        int colL = nf * 16 + (lane & 15);
        int mb = (lane >> 4) << 3;
#pragma unroll
        for (int r = 0; r < 8; ++r) {
            float ha = ca[nf].f[r], hb = cb[nf].f[r];
            float sg = 1.f / (1.f + __expf(-ha));
            float u = ha * sg;
            float sp = sg * (1.f + ha * (1.f - sg));
            float s = u * hb;
            int ml = mb + r;
            size_t gi = (size_t)(m0 + rw + ml) * 1024 + h * 64 + colL;
            U[gi] = f2b(u); V[gi] = f2b(hb); SP[gi] = f2b(sp);
            unsigned short sb = f2b(s);
            S[gi] = sb;
            ss[ml * 72 + colL] = sb;   // wave-private stage for X = s@w3
        }
    }
    FragC cx[4];
#pragma unroll
    for (int nf = 0; nf < 4; ++nf)
#pragma unroll
        for (int r = 0; r < 8; ++r) cx[nf].f[r] = 0.f;
#pragma unroll
    for (int kt = 0; kt < 64; kt += 32) {
        v16bf a = load_frag_a(&ss[kt], 72);
        int kb = (kt >> 5) << 2;
#pragma unroll
        for (int nf = 0; nf < 4; ++nf)
            cx[nf].v = wmma_bf16(a, load_frag_packed(&w3s[(kb + nf) << 9]), cx[nf].v);
    }
#pragma unroll
    for (int nf = 0; nf < 4; ++nf) {
        int colL = nf * 16 + (lane & 15);
        int mb = (lane >> 4) << 3;
#pragma unroll
        for (int r = 0; r < 8; ++r) {
            size_t gi = (size_t)(m0 + rw + mb + r) * 1024 + h * 64 + colL;
            X[gi] = f2b(cx[nf].f[r]);
        }
    }
}

// ---------------- per-head GEMM: Out[n,h*64+j] = sum_k A[n,h*64+k]*W[h][k][j] ----------------
__global__ __launch_bounds__(256)
void head_gemm_kernel(const unsigned short* __restrict__ A,
                      const unsigned short* __restrict__ W,
                      unsigned short* __restrict__ Out) {
    __shared__ __align__(16) unsigned short as[128 * 72];
    __shared__ __align__(16) unsigned short wl[8 * 512];
    int tid = threadIdx.x, h = blockIdx.x, m0 = blockIdx.y * 128;
    stage_act_tile(A + (size_t)m0 * 1024 + h * 64, as);
    stage_weight_packed(W + h * 4096, wl);
    __syncthreads();
    int w = tid >> 5, lane = tid & 31, rw = w * 16;
    FragC c[4];
#pragma unroll
    for (int nf = 0; nf < 4; ++nf)
#pragma unroll
        for (int r = 0; r < 8; ++r) c[nf].f[r] = 0.f;
#pragma unroll
    for (int kt = 0; kt < 64; kt += 32) {
        v16bf a = load_frag_a(&as[rw * 72 + kt], 72);
        int kb = (kt >> 5) << 2;
#pragma unroll
        for (int nf = 0; nf < 4; ++nf)
            c[nf].v = wmma_bf16(a, load_frag_packed(&wl[(kb + nf) << 9]), c[nf].v);
    }
#pragma unroll
    for (int nf = 0; nf < 4; ++nf) {
        int colL = nf * 16 + (lane & 15);
        int mb = (lane >> 4) << 3;
#pragma unroll
        for (int r = 0; r < 8; ++r) {
            size_t gi = (size_t)(m0 + rw + mb + r) * 1024 + h * 64 + colL;
            Out[gi] = f2b(c[nf].f[r]);
        }
    }
}

// ---------------- split-K grad GEMM: G[h][d][f] += sum_n A[n,h*64+d]*B[n,h*64+f] ----------------
__global__ __launch_bounds__(256)
void grad_gemm_kernel(const unsigned short* __restrict__ Abuf,
                      const unsigned short* __restrict__ Bbuf,
                      float* __restrict__ Gout) {
    __shared__ __align__(16) unsigned short at[4 * 512];  // packed A^T (M=64, K=32)
    __shared__ __align__(16) unsigned short bt[4 * 512];  // packed B   (K=32, N=64)
    int h = blockIdx.x;
    int n0 = blockIdx.y * 512;
    int tid = threadIdx.x, w = tid >> 5, lane = tid & 31;
    int d0 = (w >> 1) * 16, f0 = (w & 1) * 32;
    int gk = tid >> 4, gd = (tid & 15) << 2;      // staging: rows gk, gk+16; cols gd..gd+3
    FragC c0, c1;
#pragma unroll
    for (int r = 0; r < 8; ++r) { c0.f[r] = 0.f; c1.f[r] = 0.f; }
    for (int nc = 0; nc < 512; nc += 32) {
#pragma unroll
        for (int i = 0; i < 2; ++i) {
            int k = gk + (i << 4);
            uint2 ga = *(const uint2*)(Abuf + (size_t)(n0 + nc + k) * 1024 + h * 64 + gd);
            uint2 gb = *(const uint2*)(Bbuf + (size_t)(n0 + nc + k) * 1024 + h * 64 + gd);
            // packed A^T: m=gd+t, k: region=(gd>>4); lane=(m&15)+((k>>4)&1)<<4; pos=k&15
            unsigned short* da = &at[((gd >> 4) << 9) +
                                     (((gd & 15) + (((k >> 4) & 1) << 4)) << 4) +
                                     (k & 15)];
            da[0]  = (unsigned short)(ga.x & 0xffff);
            da[16] = (unsigned short)(ga.x >> 16);
            da[32] = (unsigned short)(ga.y & 0xffff);
            da[48] = (unsigned short)(ga.y >> 16);
            // packed B: k, n=gd+t
            unsigned short* db = &bt[((gd >> 4) << 9) + (((i << 4) + (gd & 15)) << 4) + gk];
            db[0]  = (unsigned short)(gb.x & 0xffff);
            db[16] = (unsigned short)(gb.x >> 16);
            db[32] = (unsigned short)(gb.y & 0xffff);
            db[48] = (unsigned short)(gb.y >> 16);
        }
        __syncthreads();
        v16bf a  = load_frag_packed(&at[(d0 >> 4) << 9]);
        c0.v = wmma_bf16(a, load_frag_packed(&bt[(f0 >> 4) << 9]), c0.v);
        c1.v = wmma_bf16(a, load_frag_packed(&bt[((f0 >> 4) + 1) << 9]), c1.v);
        __syncthreads();
    }
    int fA = f0 + (lane & 15);
    int db = d0 + ((lane >> 4) << 3);
#pragma unroll
    for (int r = 0; r < 8; ++r) {
        atomicAdd(&Gout[h * 4096 + (db + r) * 64 + fA], c0.f[r]);
        atomicAdd(&Gout[h * 4096 + (db + r) * 64 + fA + 16], c1.f[r]);
    }
}

// ---------------- row-wise kernels ----------------
__global__ __launch_bounds__(256)
void rms_kernel(const unsigned short* __restrict__ X,
                unsigned short* __restrict__ Y, float* __restrict__ R) {
    __shared__ float red[256];
    int row = blockIdx.x;
    const unsigned short* x = X + (size_t)row * 1024;
    float s = 0.f;
    for (int i = threadIdx.x; i < 1024; i += 256) { float v = b2f(x[i]); s += v * v; }
    red[threadIdx.x] = s; __syncthreads();
    for (int st = 128; st > 0; st >>= 1) {
        if (threadIdx.x < st) red[threadIdx.x] += red[threadIdx.x + st];
        __syncthreads();
    }
    float r = rsqrtf(red[0] * (1.f / 1024.f) + 1e-6f);
    if (threadIdx.x == 0) R[row] = r;
    unsigned short* y = Y + (size_t)row * 1024;
    for (int i = threadIdx.x; i < 1024; i += 256) y[i] = f2b(b2f(x[i]) * r);
}

__global__ __launch_bounds__(256)
void l2grad_kernel(const float* __restrict__ Pr, const float* __restrict__ PP,
                   unsigned short* __restrict__ DP) {
    __shared__ float r1[256], r2[256], r3[256];
    int row = blockIdx.x;
    const float* p = Pr + (size_t)row * 1024;
    const float* t = PP + (size_t)row * 1024;
    float a = 0.f, b = 0.f, d = 0.f;
    for (int i = threadIdx.x; i < 1024; i += 256) {
        float pv = p[i], tv = t[i];
        a += pv * pv; b += tv * tv; d += pv * tv;
    }
    r1[threadIdx.x] = a; r2[threadIdx.x] = b; r3[threadIdx.x] = d; __syncthreads();
    for (int st = 128; st > 0; st >>= 1) {
        if (threadIdx.x < st) {
            r1[threadIdx.x] += r1[threadIdx.x + st];
            r2[threadIdx.x] += r2[threadIdx.x + st];
            r3[threadIdx.x] += r3[threadIdx.x + st];
        }
        __syncthreads();
    }
    float np = fmaxf(sqrtf(r1[0]), 1e-12f);
    float nt = fmaxf(sqrtf(r2[0]), 1e-12f);
    float inp = 1.f / np;
    float k2 = (r3[0] / nt) * inp * inp * inp;
    unsigned short* dp = DP + (size_t)row * 1024;
    for (int i = threadIdx.x; i < 1024; i += 256)
        dp[i] = f2b(t[i] / nt * inp - p[i] * k2);
}

__global__ __launch_bounds__(256)
void rmsbwd_kernel(const unsigned short* __restrict__ Y, const float* __restrict__ R,
                   unsigned short* __restrict__ G) {
    __shared__ float red[256];
    int row = blockIdx.x;
    const unsigned short* y = Y + (size_t)row * 1024;
    unsigned short* g = G + (size_t)row * 1024;
    float s = 0.f;
    for (int i = threadIdx.x; i < 1024; i += 256) s += b2f(y[i]) * b2f(g[i]);
    red[threadIdx.x] = s; __syncthreads();
    for (int st = 128; st > 0; st >>= 1) {
        if (threadIdx.x < st) red[threadIdx.x] += red[threadIdx.x + st];
        __syncthreads();
    }
    float r = R[row];
    float k = red[0] * (1.f / 1024.f);
    for (int i = threadIdx.x; i < 1024; i += 256) {
        float gv = b2f(g[i]);
        g[i] = f2b(r * (gv - b2f(y[i]) * k));
    }
}

__global__ void backmul_kernel(const unsigned short* __restrict__ DS,
                               const unsigned short* __restrict__ U,
                               const unsigned short* __restrict__ V,
                               const unsigned short* __restrict__ SP,
                               unsigned short* __restrict__ DHA,
                               unsigned short* __restrict__ DV, int n) {
    int i = blockIdx.x * 256 + threadIdx.x;
    if (i >= n) return;
    float ds = b2f(DS[i]);
    float u = b2f(U[i]);
    DHA[i] = f2b(ds * b2f(V[i]) * b2f(SP[i]));
    DV[i] = f2b(ds * u);
}

__global__ void context_kernel(const float* __restrict__ Q, float* __restrict__ C) {
    int i = blockIdx.x * 256 + threadIdx.x;
    if (i >= 8 * 1024) return;
    int b = i >> 10, col = i & 1023;
    const float* q = Q + (size_t)b * 4096 * 1024 + col;
    float s = 0.f;
    for (int t = 0; t < 4096; ++t) s += q[(size_t)t * 1024];
    C[i] = s * (1.f / 4096.f);
}

__global__ void lralpha_kernel(const float* __restrict__ C,
                               const float* __restrict__ Wlr, const float* __restrict__ blr,
                               const float* __restrict__ Wd, const float* __restrict__ bd,
                               float* __restrict__ out) {
    int j = threadIdx.x;
    if (j >= 64) return;
    if (j < 48) {
        float acc = 0.f;
        for (int b = 0; b < 8; ++b) {
            float d = blr[j];
            for (int i = 0; i < 1024; ++i) d += C[b * 1024 + i] * Wlr[j * 1024 + i];
            acc += (d > 20.f) ? d : log1pf(__expf(d));
        }
        out[j] = acc * 0.125f;
    } else {
        int h = j - 48;
        float acc = 0.f;
        for (int b = 0; b < 8; ++b) {
            float d = bd[h];
            for (int i = 0; i < 1024; ++i) d += C[b * 1024 + i] * Wd[h * 1024 + i];
            acc += 1.f / (1.f + __expf(-d));
        }
        out[j] = 1.f - 0.1f * (acc * 0.125f);
    }
}

__global__ void update_kernel(const float* __restrict__ w1, const float* __restrict__ w2,
                              const float* __restrict__ w3,
                              const float* __restrict__ g1, const float* __restrict__ g2,
                              const float* __restrict__ g3,
                              const float* __restrict__ la, float* __restrict__ out) {
    int i = blockIdx.x * 256 + threadIdx.x;
    const int per = NH * HD * HD;
    if (i >= 3 * per) return;
    int which = i / per, rem = i % per, h = rem >> 12;
    const float* w = (which == 0) ? w1 : (which == 1) ? w2 : w3;
    const float* g = (which == 0) ? g1 : (which == 1) ? g2 : g3;
    float lr = la[h * 3 + which];
    float alpha = la[48 + h];
    out[33554432 + i] = alpha * w[rem] + lr * g[rem];
}

// ---------------- launcher ----------------
extern "C" void kernel_launch(void* const* d_in, const int* in_sizes, int n_in,
                              void* d_out, int out_size, void* d_ws, size_t ws_size,
                              hipStream_t stream) {
    (void)in_sizes; (void)n_in; (void)out_size; (void)ws_size;
    const float* query = (const float*)d_in[0];
    const float* post  = (const float*)d_in[1];
    const float* Wq  = (const float*)d_in[2];
    const float* bq  = (const float*)d_in[3];
    const float* Wo  = (const float*)d_in[4];
    const float* bo  = (const float*)d_in[5];
    const float* Wlr = (const float*)d_in[6];
    const float* blr = (const float*)d_in[7];
    const float* Wd  = (const float*)d_in[8];
    const float* bd  = (const float*)d_in[9];
    const float* w1  = (const float*)d_in[10];
    const float* w2  = (const float*)d_in[11];
    const float* w3  = (const float*)d_in[12];
    float* out = (float*)d_out;

    char* ws = (char*)d_ws; size_t off = 0;
    auto alloc = [&](size_t bytes) -> void* {
        void* p = ws + off; off = (off + bytes + 255) & ~(size_t)255; return p;
    };
    const size_t BIGB = NELEM * 2;                         // 64 MiB bf16 buffer
    unsigned short* P   = (unsigned short*)alloc(BIGB);    // pq bf16
    unsigned short* Ub  = (unsigned short*)alloc(BIGB);    // u   -> later dv
    unsigned short* Vb  = (unsigned short*)alloc(BIGB);    // v
    unsigned short* SPb = (unsigned short*)alloc(BIGB);    // silu'(ha)
    unsigned short* Sb  = (unsigned short*)alloc(BIGB);    // s
    unsigned short* Yb  = (unsigned short*)alloc(BIGB);    // y   -> later ds
    unsigned short* T1  = (unsigned short*)alloc(BIGB);    // query_bf16 -> dprior -> dha
    unsigned short* T2  = (unsigned short*)alloc(BIGB);    // X -> dY -> dX
    unsigned short* WqT  = (unsigned short*)alloc((size_t)NIN * NOUT * 2);
    unsigned short* Wob  = (unsigned short*)alloc((size_t)NOUT * NOUT * 2);
    unsigned short* WoTb = (unsigned short*)alloc((size_t)NOUT * NOUT * 2);
    unsigned short* w1b  = (unsigned short*)alloc(NH * HD * HD * 2);
    unsigned short* w2b  = (unsigned short*)alloc(NH * HD * HD * 2);
    unsigned short* w3b  = (unsigned short*)alloc(NH * HD * HD * 2);
    unsigned short* w3Tb = (unsigned short*)alloc(NH * HD * HD * 2);
    float* g1 = (float*)alloc(NH * HD * HD * 4);
    float* g2 = (float*)alloc(NH * HD * HD * 4);
    float* g3 = (float*)alloc(NH * HD * HD * 4);
    float* Rr  = (float*)alloc(NBT * 4);
    float* Ctx = (float*)alloc(8 * 1024 * 4);
    float* LA  = (float*)alloc(64 * 4);

    // ---- operand conversion ----
    convert_kernel<<<(int)(NELEM / 256), 256, 0, stream>>>(query, T1, (int)NELEM);
    transpose_convert_kernel<<<dim3(4096, 1), 256, 0, stream>>>(Wq, WqT, 1024, 1024);
    convert_kernel<<<4096, 256, 0, stream>>>(Wo, Wob, 1024 * 1024);
    transpose_convert_kernel<<<dim3(4096, 1), 256, 0, stream>>>(Wo, WoTb, 1024, 1024);
    convert_kernel<<<256, 256, 0, stream>>>(w1, w1b, NH * HD * HD);
    convert_kernel<<<256, 256, 0, stream>>>(w2, w2b, NH * HD * HD);
    convert_kernel<<<256, 256, 0, stream>>>(w3, w3b, NH * HD * HD);
    transpose_convert_kernel<<<dim3(16, NH), 256, 0, stream>>>(w3, w3Tb, HD, HD);

    // ---- forward ----
    gemm_bf16_128x64<<<dim3(16, 256), 256, 0, stream>>>(T1, WqT, bq, nullptr, P,
                                                        NBT, NOUT, NIN);       // pq
    head_forward_kernel<<<dim3(NH, 256), 256, 0, stream>>>(P, w1b, w2b, w3b,
                                                           Ub, Vb, SPb, Sb, T2); // X
    rms_kernel<<<NBT, 256, 0, stream>>>(T2, Yb, Rr);
    gemm_bf16_128x64<<<dim3(16, 256), 256, 0, stream>>>(Yb, WoTb, bo, out, nullptr,
                                                        NBT, NOUT, NOUT);      // read_out

    // ---- backward ----
    l2grad_kernel<<<NBT, 256, 0, stream>>>(out, post, T1);                     // dprior
    gemm_bf16_128x64<<<dim3(16, 256), 256, 0, stream>>>(T1, Wob, nullptr, nullptr, T2,
                                                        NBT, NOUT, NOUT);      // dY
    rmsbwd_kernel<<<NBT, 256, 0, stream>>>(Yb, Rr, T2);                        // dX (in place)
    head_gemm_kernel<<<dim3(NH, 256), 256, 0, stream>>>(T2, w3Tb, Yb);         // ds
    backmul_kernel<<<(int)(NELEM / 256), 256, 0, stream>>>(Yb, Ub, Vb, SPb,
                                                           T1, Ub, (int)NELEM); // dha, dv
    zerof_kernel<<<768, 256, 0, stream>>>(g1, NH * HD * HD);
    zerof_kernel<<<768, 256, 0, stream>>>(g2, NH * HD * HD);
    zerof_kernel<<<768, 256, 0, stream>>>(g3, NH * HD * HD);
    grad_gemm_kernel<<<dim3(NH, 64), 256, 0, stream>>>(P, T1, g1);             // g1 = pq^T dha
    grad_gemm_kernel<<<dim3(NH, 64), 256, 0, stream>>>(P, Ub, g2);             // g2 = pq^T dv
    grad_gemm_kernel<<<dim3(NH, 64), 256, 0, stream>>>(Sb, T2, g3);            // g3 = s^T dX

    // ---- gates + weight update ----
    context_kernel<<<32, 256, 0, stream>>>(query, Ctx);
    lralpha_kernel<<<1, 64, 0, stream>>>(Ctx, Wlr, blr, Wd, bd, LA);
    update_kernel<<<768, 256, 0, stream>>>(w1, w2, w3, g1, g2, g3, LA, out);
}
